// MultiHeadAttention_73624329388477
// MI455X (gfx1250) — matrix operand order repaired and checked by reading, not compiled, not run on hardware
//
#include <hip/hip_runtime.h>
#include <hip/hip_bf16.h>

// ---------------------------------------------------------------------------
// MHA forward for MI455X (gfx1250): bf16 WMMA GEMMs + fused softmax.
// d_out = [ out: B*S*D f32 | attn: B*H*S*S f32 ]
// ---------------------------------------------------------------------------

typedef __bf16 bf16_t;
typedef __attribute__((ext_vector_type(16))) __bf16 v16bf;
typedef __attribute__((ext_vector_type(8)))  __bf16 v8bf;
typedef __attribute__((ext_vector_type(8)))  float  v8f;

#define Bq  8
#define Sq  1024
#define Dq  1024
#define Hq  16
#define DEPTH 64
#define Mrows (Bq * Sq)   // 8192

__device__ __forceinline__ v8f wmma_bf16(v16bf a, v16bf b, v8f c) {
    // 8 args: (neg_a, A, neg_b, B, c_mod, C, reuse_a, reuse_b)
    return __builtin_amdgcn_wmma_f32_16x16x32_bf16(false, a, false, b, (short)0, c,
                                                   false, false);
}

// A fragment (16x32 bf16, M x K): lane holds row m = lane%16,
// K chunks [half*8 .. +7] and [16+half*8 .. +7] contiguous in row-major src.
__device__ __forceinline__ v16bf load_a_bf(const bf16_t* __restrict__ base, int ld,
                                           int mbase, int kk, int lane) {
    const int m    = mbase + (lane & 15);
    const int half = lane >> 4;
    const bf16_t* rp = base + (size_t)m * ld + kk + half * 8;
    v8bf lo = *(const v8bf*)rp;
    v8bf hi = *(const v8bf*)(rp + 16);
    v16bf a;
#pragma unroll
    for (int i = 0; i < 8; ++i) { a[i] = lo[i]; a[8 + i] = hi[i]; }
    return a;
}

// Same A fragment but converting from f32 source on the fly.
__device__ __forceinline__ v16bf load_a_f32(const float* __restrict__ base, int ld,
                                            int mbase, int kk, int lane) {
    const int m    = mbase + (lane & 15);
    const int half = lane >> 4;
    const float* rp = base + (size_t)m * ld + kk + half * 8;
    v16bf a;
#pragma unroll
    for (int i = 0; i < 8; ++i) {
        a[i]     = (bf16_t)rp[i];
        a[8 + i] = (bf16_t)rp[16 + i];
    }
    return a;
}

// B fragment (32x16 bf16, K x N) from a row-major [N][K] (i.e. transposed) src:
// lane holds column n = lane%16, K chunks [half*8 .. +7] and [16+half*8 .. +7].
__device__ __forceinline__ v16bf load_b_bf(const bf16_t* __restrict__ baseT, int ld,
                                           int nbase, int kk, int lane) {
    const int n    = nbase + (lane & 15);
    const int half = lane >> 4;
    const bf16_t* rp = baseT + (size_t)n * ld + kk + half * 8;
    v8bf lo = *(const v8bf*)rp;
    v8bf hi = *(const v8bf*)(rp + 16);
    v16bf b;
#pragma unroll
    for (int i = 0; i < 8; ++i) { b[i] = lo[i]; b[8 + i] = hi[i]; }
    return b;
}

// ---------------------------------------------------------------------------
// Kernel 0: f32 -> bf16 convert
// ---------------------------------------------------------------------------
__global__ void cvt_f32_to_bf16(const float* __restrict__ x, bf16_t* __restrict__ y,
                                int n) {
    int i = blockIdx.x * blockDim.x + threadIdx.x;
    if (i < n) y[i] = (bf16_t)x[i];
}

// ---------------------------------------------------------------------------
// Kernel 1: W[K=1024][N=1024] f32 -> WT[N][K] bf16  (16x16 LDS tile transpose)
// ---------------------------------------------------------------------------
__global__ __launch_bounds__(256) void transpose_w(const float* __restrict__ w,
                                                   bf16_t* __restrict__ wt) {
    __shared__ float tile[16][17];
    const int tx = threadIdx.x & 15;
    const int ty = threadIdx.x >> 4;
    const int k0 = blockIdx.x * 16;
    const int n0 = blockIdx.y * 16;
    tile[ty][tx] = w[(size_t)(k0 + ty) * Dq + n0 + tx];
    __syncthreads();
    wt[(size_t)(n0 + ty) * Dq + k0 + tx] = (bf16_t)tile[tx][ty];
}

// ---------------------------------------------------------------------------
// Kernel 2: projection GEMM  C = relu(X @ W + b) -> BatchNorm -> head-split bf16
//   mode 0: qh [B,H,S,64] (pre-scaled by 1/sqrt(64))
//   mode 1: kh [B,H,S,64]
//   mode 2: vhT [B,H,64,S]  (transposed so attn@V B-frags are contiguous)
// WG = 256 threads (8 waves); WG tile 128x64; wave tile 16x64 (4 accum frags).
// B-frag loads batched ahead of the WMMA group for memory-level parallelism.
// ---------------------------------------------------------------------------
__global__ __launch_bounds__(256) void proj_gemm(
    const bf16_t* __restrict__ X, const bf16_t* __restrict__ WT,
    const float* __restrict__ bias,
    const float* __restrict__ gam, const float* __restrict__ bet,
    const float* __restrict__ mmean, const float* __restrict__ mvar,
    bf16_t* __restrict__ out, int mode) {
    const int lane  = threadIdx.x & 31;
    const int wave  = threadIdx.x >> 5;
    const int mbase = blockIdx.x * 128 + wave * 16;
    const int nbase = blockIdx.y * 64;

    v8f acc[4] = {};
    for (int kk = 0; kk < Dq; kk += 32) {
        __builtin_prefetch(X + (size_t)(mbase + (lane & 15)) * Dq + kk + 32, 0, 1);
        v16bf a = load_a_bf(X, Dq, mbase, kk, lane);
        v16bf bfr[4];
#pragma unroll
        for (int t = 0; t < 4; ++t)
            bfr[t] = load_b_bf(WT, Dq, nbase + t * 16, kk, lane);
#pragma unroll
        for (int t = 0; t < 4; ++t)
            acc[t] = wmma_bf16(a, bfr[t], acc[t]);
    }

    const int half = lane >> 4, nl = lane & 15;
#pragma unroll
    for (int t = 0; t < 4; ++t) {
        const int n = nbase + t * 16 + nl;
        const float sc = gam[n] * rsqrtf(mvar[n] + 1e-3f);
        const float sh = bet[n] - mmean[n] * sc;
        const float bs = bias[n];
        const int hh = n >> 6, dd = n & 63;
#pragma unroll
        for (int r = 0; r < 8; ++r) {
            const int m  = mbase + half * 8 + r;
            const int bb = m >> 10, ss = m & 1023;
            float v = fmaxf(acc[t][r] + bs, 0.0f);
            v = v * sc + sh;
            if (mode == 2) {
                // vhT[b,h,d,s]
                out[(((size_t)(bb * Hq + hh) * DEPTH + dd) << 10) + ss] = (bf16_t)v;
            } else {
                if (mode == 0) v *= 0.125f;  // fold 1/sqrt(depth) into qh
                // [b,h,s,d]
                out[(((size_t)(bb * Hq + hh) * Sq + ss) << 6) + dd] = (bf16_t)v;
            }
        }
    }
}

// ---------------------------------------------------------------------------
// Kernel 3: scores + mask + softmax -> attn f32 (coalesced via LDS transpose)
// One WG per (bh, 16-query-row block). 8 waves x 8 tiles cover 16x1024 scores.
// ---------------------------------------------------------------------------
__global__ __launch_bounds__(256) void attn_softmax(
    const bf16_t* __restrict__ qh, const bf16_t* __restrict__ kh,
    const float* __restrict__ mask, float* __restrict__ attn) {
    const int lane  = threadIdx.x & 31;
    const int wave  = threadIdx.x >> 5;
    const int bh    = blockIdx.y;           // b*16 + h
    const int b     = bh >> 4;
    const int mbase = blockIdx.x * 16;      // query-row block within S

    const bf16_t* qp = qh + (size_t)bh * Sq * DEPTH;
    const bf16_t* kp = kh + (size_t)bh * Sq * DEPTH;
    const float*  mp = mask + (size_t)b * Sq;
    float*        ap = attn + (size_t)bh * Sq * Sq;

    v8f acc[8] = {};
#pragma unroll
    for (int kk = 0; kk < DEPTH; kk += 32) {
        v16bf a = load_a_bf(qp, DEPTH, mbase, kk, lane);
#pragma unroll
        for (int g = 0; g < 2; ++g) {
            v16bf bfr[4];
#pragma unroll
            for (int t = 0; t < 4; ++t)
                bfr[t] = load_b_bf(kp, DEPTH, wave * 128 + (g * 4 + t) * 16, kk, lane);
#pragma unroll
            for (int t = 0; t < 4; ++t)
                acc[g * 4 + t] = wmma_bf16(a, bfr[t], acc[g * 4 + t]);
        }
    }

    const int half = lane >> 4, nl = lane & 15;
    // mask (column-indexed)
#pragma unroll
    for (int t = 0; t < 8; ++t) {
        const float madd = mp[wave * 128 + t * 16 + nl] * -1e9f;
#pragma unroll
        for (int r = 0; r < 8; ++r) acc[t][r] += madd;
    }

    __shared__ float red[8][16];
    __shared__ float probs[8][1024];  // 32 KB staging for coalesced writes

    // ---- row max ----
    float rmax[8];
#pragma unroll
    for (int r = 0; r < 8; ++r) {
        float v = acc[0][r];
#pragma unroll
        for (int t = 1; t < 8; ++t) v = fmaxf(v, acc[t][r]);
        for (int off = 1; off < 16; off <<= 1) v = fmaxf(v, __shfl_xor(v, off, 32));
        rmax[r] = v;
    }
    if (nl == 0) {
#pragma unroll
        for (int r = 0; r < 8; ++r) red[wave][half * 8 + r] = rmax[r];
    }
    __syncthreads();
#pragma unroll
    for (int r = 0; r < 8; ++r) {
        float v = red[0][half * 8 + r];
        for (int w2 = 1; w2 < 8; ++w2) v = fmaxf(v, red[w2][half * 8 + r]);
        rmax[r] = v;
    }
    __syncthreads();

    // ---- exp + row sum ----
    float rinv[8];
#pragma unroll
    for (int r = 0; r < 8; ++r) {
        float s = 0.0f;
#pragma unroll
        for (int t = 0; t < 8; ++t) {
            acc[t][r] = __expf(acc[t][r] - rmax[r]);
            s += acc[t][r];
        }
        for (int off = 1; off < 16; off <<= 1) s += __shfl_xor(s, off, 32);
        rinv[r] = s;
    }
    if (nl == 0) {
#pragma unroll
        for (int r = 0; r < 8; ++r) red[wave][half * 8 + r] = rinv[r];
    }
    __syncthreads();
#pragma unroll
    for (int r = 0; r < 8; ++r) {
        float s = red[0][half * 8 + r];
        for (int w2 = 1; w2 < 8; ++w2) s += red[w2][half * 8 + r];
        rinv[r] = 1.0f / s;
    }
    __syncthreads();

    // ---- normalize, stage 8 rows at a time in LDS, write coalesced ----
    for (int ph = 0; ph < 2; ++ph) {
        if (half == ph) {
#pragma unroll
            for (int t = 0; t < 8; ++t) {
#pragma unroll
                for (int r = 0; r < 8; ++r)
                    probs[r][wave * 128 + t * 16 + nl] = acc[t][r] * rinv[r];
            }
        }
        __syncthreads();
        for (int i = threadIdx.x; i < 8 * 1024; i += 256) {
            const int row = i >> 10, col = i & 1023;
            ap[(size_t)(mbase + ph * 8 + row) * Sq + col] = probs[row][col];
        }
        __syncthreads();
    }
}

// ---------------------------------------------------------------------------
// Kernel 4: ctx = attn @ V   (per (b,h): [1024x1024] @ [1024x64])
// A read from attn f32 (converted to bf16), B from vhT bf16 (contiguous).
// ctx stored bf16 merged [B*S, D] for the final projection.
// ---------------------------------------------------------------------------
__global__ __launch_bounds__(256) void ctx_gemm(
    const float* __restrict__ attn, const bf16_t* __restrict__ vhT,
    bf16_t* __restrict__ ctx) {
    const int lane  = threadIdx.x & 31;
    const int wave  = threadIdx.x >> 5;
    const int bh    = blockIdx.y;
    const int b     = bh >> 4, h = bh & 15;
    const int mbase = blockIdx.x * 128 + wave * 16;  // s row

    const float*  ap = attn + (size_t)bh * Sq * Sq;
    const bf16_t* vp = vhT + (size_t)bh * DEPTH * Sq;

    v8f acc[4] = {};
    for (int kk = 0; kk < Sq; kk += 32) {
        __builtin_prefetch(ap + (size_t)(mbase + (lane & 15)) * Sq + kk + 32, 0, 1);
        v16bf bfr[4];
#pragma unroll
        for (int t = 0; t < 4; ++t)
            bfr[t] = load_b_bf(vp, Sq, t * 16, kk, lane);
        v16bf a = load_a_f32(ap, Sq, mbase, kk, lane);
#pragma unroll
        for (int t = 0; t < 4; ++t)
            acc[t] = wmma_bf16(a, bfr[t], acc[t]);
    }

    const int half = lane >> 4, nl = lane & 15;
#pragma unroll
    for (int t = 0; t < 4; ++t) {
        const int d = t * 16 + nl;
#pragma unroll
        for (int r = 0; r < 8; ++r) {
            const int s = mbase + half * 8 + r;
            ctx[((size_t)(b * Sq + s) << 10) + h * DEPTH + d] = (bf16_t)acc[t][r];
        }
    }
}

// ---------------------------------------------------------------------------
// Kernel 5: out = relu(ctx @ wo + bo) -> f32 [B,S,D]
// ---------------------------------------------------------------------------
__global__ __launch_bounds__(256) void out_gemm(
    const bf16_t* __restrict__ X, const bf16_t* __restrict__ WT,
    const float* __restrict__ bias, float* __restrict__ out) {
    const int lane  = threadIdx.x & 31;
    const int wave  = threadIdx.x >> 5;
    const int mbase = blockIdx.x * 128 + wave * 16;
    const int nbase = blockIdx.y * 64;

    v8f acc[4] = {};
    for (int kk = 0; kk < Dq; kk += 32) {
        __builtin_prefetch(X + (size_t)(mbase + (lane & 15)) * Dq + kk + 32, 0, 1);
        v16bf a = load_a_bf(X, Dq, mbase, kk, lane);
        v16bf bfr[4];
#pragma unroll
        for (int t = 0; t < 4; ++t)
            bfr[t] = load_b_bf(WT, Dq, nbase + t * 16, kk, lane);
#pragma unroll
        for (int t = 0; t < 4; ++t)
            acc[t] = wmma_bf16(a, bfr[t], acc[t]);
    }

    const int half = lane >> 4, nl = lane & 15;
#pragma unroll
    for (int t = 0; t < 4; ++t) {
        const int n = nbase + t * 16 + nl;
        const float bs = bias[n];
#pragma unroll
        for (int r = 0; r < 8; ++r) {
            const int m = mbase + half * 8 + r;
            out[(size_t)m * Dq + n] = fmaxf(acc[t][r] + bs, 0.0f);
        }
    }
}

// ---------------------------------------------------------------------------
extern "C" void kernel_launch(void* const* d_in, const int* in_sizes, int n_in,
                              void* d_out, int out_size, void* d_ws, size_t ws_size,
                              hipStream_t stream) {
    const float* q    = (const float*)d_in[0];
    const float* k    = (const float*)d_in[1];
    const float* v    = (const float*)d_in[2];
    const float* mask = (const float*)d_in[3];
    const float* wq = (const float*)d_in[4];  const float* bq = (const float*)d_in[5];
    const float* wk = (const float*)d_in[6];  const float* bk = (const float*)d_in[7];
    const float* wv = (const float*)d_in[8];  const float* bv = (const float*)d_in[9];
    const float* wo = (const float*)d_in[10]; const float* bo = (const float*)d_in[11];
    const float* g1 = (const float*)d_in[12]; const float* be1 = (const float*)d_in[13];
    const float* mm1 = (const float*)d_in[14]; const float* mv1 = (const float*)d_in[15];
    const float* g2 = (const float*)d_in[16]; const float* be2 = (const float*)d_in[17];
    const float* mm2 = (const float*)d_in[18]; const float* mv2 = (const float*)d_in[19];
    const float* g3 = (const float*)d_in[20]; const float* be3 = (const float*)d_in[21];
    const float* mm3 = (const float*)d_in[22]; const float* mv3 = (const float*)d_in[23];

    const size_t MD = (size_t)Mrows * Dq;   // 8192*1024
    const size_t DD = (size_t)Dq * Dq;      // 1024*1024

    bf16_t* xq  = (bf16_t*)d_ws;
    bf16_t* xk  = xq + MD;
    bf16_t* xv  = xk + MD;
    bf16_t* wqT = xv + MD;
    bf16_t* wkT = wqT + DD;
    bf16_t* wvT = wkT + DD;
    bf16_t* woT = wvT + DD;
    bf16_t* qh  = woT + DD;
    bf16_t* kh  = qh + MD;
    bf16_t* vhT = kh + MD;
    bf16_t* ctx = vhT + MD;

    float* out  = (float*)d_out;
    float* attn = out + MD;  // attn region after out region

    // 0) convert activations to bf16
    {
        const int n = (int)MD, thr = 256, blk = (n + thr - 1) / thr;
        cvt_f32_to_bf16<<<blk, thr, 0, stream>>>(q, xq, n);
        cvt_f32_to_bf16<<<blk, thr, 0, stream>>>(k, xk, n);
        cvt_f32_to_bf16<<<blk, thr, 0, stream>>>(v, xv, n);
    }
    // 1) transpose+convert weights
    {
        dim3 grid(Dq / 16, Dq / 16);
        transpose_w<<<grid, 256, 0, stream>>>(wq, wqT);
        transpose_w<<<grid, 256, 0, stream>>>(wk, wkT);
        transpose_w<<<grid, 256, 0, stream>>>(wv, wvT);
        transpose_w<<<grid, 256, 0, stream>>>(wo, woT);
    }
    // 2) projections (fused bias+relu+BN, head split)
    {
        dim3 grid(Mrows / 128, Dq / 64);
        proj_gemm<<<grid, 256, 0, stream>>>(xq, wqT, bq, g1, be1, mm1, mv1, qh, 0);
        proj_gemm<<<grid, 256, 0, stream>>>(xk, wkT, bk, g2, be2, mm2, mv2, kh, 1);
        proj_gemm<<<grid, 256, 0, stream>>>(xv, wvT, bv, g3, be3, mm3, mv3, vhT, 2);
    }
    // 3) scores + mask + softmax -> attn (f32, written once, coalesced)
    attn_softmax<<<dim3(Sq / 16, Bq * Hq), 256, 0, stream>>>(qh, kh, mask, attn);
    // 4) ctx = attn @ V
    ctx_gemm<<<dim3(Sq / 128, Bq * Hq), 256, 0, stream>>>(attn, vhT, ctx);
    // 5) out = relu(ctx @ wo + bo)
    out_gemm<<<dim3(Mrows / 128, Dq / 64), 256, 0, stream>>>(ctx, woT, bo, out);
}
